// Model_pool_13511967113352
// MI455X (gfx1250) — matrix operand order repaired and checked by reading, not compiled, not run on hardware
//
#include <hip/hip_runtime.h>
#include <math.h>

// ---------------------------------------------------------------------------
// Problem constants (match reference)
// ---------------------------------------------------------------------------
#define BGR 256                 // graphs
#define NND 512                 // nodes per graph
#define DEGE 16
#define NT (BGR * NND)          // 131072 total nodes
#define EE (NT * DEGE)          // 2097152 edges
#define FIN 64
#define HH 128
#define KP1 256
#define KP2 128

typedef __attribute__((ext_vector_type(16))) __bf16 v16bf;
typedef __attribute__((ext_vector_type(8)))  __bf16 v8bf;
typedef __attribute__((ext_vector_type(8)))  float  v8f;

// ---------------------------------------------------------------------------
// Wave helpers (wave32)
// ---------------------------------------------------------------------------
__device__ __forceinline__ float waveSum(float v) {
  #pragma unroll
  for (int o = 16; o > 0; o >>= 1) v += __shfl_xor(v, o, 32);
  return v;
}

__device__ __forceinline__ void atomicMaxFloat(float* addr, float val) {
  // works with -inf init: signed max for >=0, unsigned min for <0
  if (val >= 0.0f) atomicMax((int*)addr, __float_as_int(val));
  else             atomicMin((unsigned int*)addr, (unsigned int)__float_as_int(val));
}

__device__ __forceinline__ void cvt16(v16bf& v, float4 a, float4 b, float4 c, float4 d) {
  v[0]  = (__bf16)a.x; v[1]  = (__bf16)a.y; v[2]  = (__bf16)a.z; v[3]  = (__bf16)a.w;
  v[4]  = (__bf16)b.x; v[5]  = (__bf16)b.y; v[6]  = (__bf16)b.z; v[7]  = (__bf16)b.w;
  v[8]  = (__bf16)c.x; v[9]  = (__bf16)c.y; v[10] = (__bf16)c.z; v[11] = (__bf16)c.w;
  v[12] = (__bf16)d.x; v[13] = (__bf16)d.y; v[14] = (__bf16)d.z; v[15] = (__bf16)d.w;
}

__device__ __forceinline__ v16bf pack16(v8bf lo, v8bf hi) {
  v16bf a;
  #pragma unroll
  for (int i = 0; i < 8; ++i) { a[i] = lo[i]; a[8 + i] = hi[i]; }
  return a;
}

// ---------------------------------------------------------------------------
// WMMA GEMM (f32 A): C[b] = act( A[b] x B[b] + bias + Cadd )
//   A: [M,K] f32 row-major; B: [K,N] f32 row-major.
//   One wave = 32x32 C macro-tile (2x2 of 16x16), K-loop of bf16 16x16x32.
//   Requires M % 32 == 0 and K % 32 == 0 (true for all call sites).
// Fragment layouts per CDNA5 ISA 7.12.2 (wave32):
//   A 16x32 bf16 : lanes 0-15 -> M=lane, kb=0 ; lanes 16-31 -> kb=8
//                  elem i<8: K=kb+i ; i>=8: K=16+kb+(i-8)  (two 8-elem runs)
//   B 32x16 bf16 : col = n0 + (lane&15); element i -> K row, same k order as A
//   C 16x16 f32  : vgpr r -> M = r + 8*(lane>>4), col = n0 + (lane&15)
// ---------------------------------------------------------------------------
__global__ __launch_bounds__(128)
void wmma_gemm_f32a_kernel(const float* __restrict__ A, const float* __restrict__ Bw,
                           const float* __restrict__ bias, const float* __restrict__ Cadd,
                           float* __restrict__ C,
                           int M, int N, int K,
                           long long sA, long long sB, long long sC,
                           int lda, int ldb, int ldc, int doRelu)
{
  const int wave = threadIdx.x >> 5;
  const int lane = threadIdx.x & 31;
  const int m0 = blockIdx.x * 32;
  const int n0 = blockIdx.y * 128 + wave * 32;
  if (m0 >= M || n0 >= N) return;          // wave-uniform: EXEC stays all-ones

  A  += (long long)blockIdx.z * sA;
  Bw += (long long)blockIdx.z * sB;
  C  += (long long)blockIdx.z * sC;

  const int l15 = lane & 15;
  const int kb  = (lane >> 4) * 8;

  v8f acc00 = {}, acc01 = {}, acc10 = {}, acc11 = {};

  const float* Bl  = Bw + (size_t)kb * ldb + n0 + l15;        // K row = k0+kb, col n0+l15
  const float* Ap0 = A + (size_t)(m0 + l15) * lda + kb;
  const float* Ap1 = Ap0 + (size_t)16 * lda;

  for (int k0 = 0; k0 < K; k0 += 32) {
    v16bf a0, a1, b0, b1;

    float4 x0 = *(const float4*)(Ap0 +  0);
    float4 x1 = *(const float4*)(Ap0 +  4);
    float4 x2 = *(const float4*)(Ap0 + 16);
    float4 x3 = *(const float4*)(Ap0 + 20);
    cvt16(a0, x0, x1, x2, x3);
    float4 y0 = *(const float4*)(Ap1 +  0);
    float4 y1 = *(const float4*)(Ap1 +  4);
    float4 y2 = *(const float4*)(Ap1 + 16);
    float4 y3 = *(const float4*)(Ap1 + 20);
    cvt16(a1, y0, y1, y2, y3);
    Ap0 += 32; Ap1 += 32;

    #pragma unroll
    for (int i = 0; i < 8; ++i) {
      b0[i]     = (__bf16)Bl[(size_t)i * ldb];
      b0[8 + i] = (__bf16)Bl[(size_t)(16 + i) * ldb];
      b1[i]     = (__bf16)Bl[(size_t)i * ldb + 16];
      b1[8 + i] = (__bf16)Bl[(size_t)(16 + i) * ldb + 16];
    }
    if (k0 + 32 < K)   // stream next B panel (global_prefetch_b8)
      __builtin_prefetch(Bl + (size_t)32 * ldb, 0, 1);
    Bl += (size_t)32 * ldb;

    acc00 = __builtin_amdgcn_wmma_f32_16x16x32_bf16(false, a0, false, b0, (short)0, acc00, false, false);
    acc01 = __builtin_amdgcn_wmma_f32_16x16x32_bf16(false, a0, false, b1, (short)0, acc01, false, false);
    acc10 = __builtin_amdgcn_wmma_f32_16x16x32_bf16(false, a1, false, b0, (short)0, acc10, false, false);
    acc11 = __builtin_amdgcn_wmma_f32_16x16x32_bf16(false, a1, false, b1, (short)0, acc11, false, false);
  }

  const int rowOff = 8 * (lane >> 4);
  auto storeTile = [&](const v8f& acc, int mt, int nt) {
    const int col = n0 + nt * 16 + l15;
    const float bv = bias ? bias[col] : 0.0f;
    #pragma unroll
    for (int r = 0; r < 8; ++r) {
      const int mrow = m0 + mt * 16 + r + rowOff;
      const long long idx = (long long)mrow * ldc + col;
      float v = acc[r] + bv;
      if (Cadd) v += Cadd[idx];
      if (doRelu) v = fmaxf(v, 0.0f);
      C[idx] = v;
    }
  };
  storeTile(acc00, 0, 0);
  storeTile(acc01, 0, 1);
  storeTile(acc10, 1, 0);
  storeTile(acc11, 1, 1);
}

// ---------------------------------------------------------------------------
// WMMA GEMM (bf16 A, pre-transposed): A fragments are two contiguous 16-byte
// bf16 loads per lane, already in WMMA register format (no cvt in the loop).
// Used for the pooling einsums s^T x / s^T A with A = transposed softmax.
// ---------------------------------------------------------------------------
__global__ __launch_bounds__(128)
void wmma_gemm_bf16a_kernel(const __bf16* __restrict__ A, const float* __restrict__ Bw,
                            float* __restrict__ C,
                            int M, int N, int K,
                            long long sA, long long sB, long long sC,
                            int lda, int ldb, int ldc)
{
  const int wave = threadIdx.x >> 5;
  const int lane = threadIdx.x & 31;
  const int m0 = blockIdx.x * 32;
  const int n0 = blockIdx.y * 128 + wave * 32;
  if (m0 >= M || n0 >= N) return;

  A  += (long long)blockIdx.z * sA;
  Bw += (long long)blockIdx.z * sB;
  C  += (long long)blockIdx.z * sC;

  const int l15 = lane & 15;
  const int kb  = (lane >> 4) * 8;

  v8f acc00 = {}, acc01 = {}, acc10 = {}, acc11 = {};

  const float*  Bl  = Bw + (size_t)kb * ldb + n0 + l15;
  const __bf16* Ap0 = A + (size_t)(m0 + l15) * lda + kb;
  const __bf16* Ap1 = Ap0 + (size_t)16 * lda;

  for (int k0 = 0; k0 < K; k0 += 32) {
    v16bf b0, b1;
    v16bf a0 = pack16(*(const v8bf*)(Ap0), *(const v8bf*)(Ap0 + 16));
    v16bf a1 = pack16(*(const v8bf*)(Ap1), *(const v8bf*)(Ap1 + 16));
    Ap0 += 32; Ap1 += 32;

    #pragma unroll
    for (int i = 0; i < 8; ++i) {
      b0[i]     = (__bf16)Bl[(size_t)i * ldb];
      b0[8 + i] = (__bf16)Bl[(size_t)(16 + i) * ldb];
      b1[i]     = (__bf16)Bl[(size_t)i * ldb + 16];
      b1[8 + i] = (__bf16)Bl[(size_t)(16 + i) * ldb + 16];
    }
    if (k0 + 32 < K)
      __builtin_prefetch(Bl + (size_t)32 * ldb, 0, 1);
    Bl += (size_t)32 * ldb;

    acc00 = __builtin_amdgcn_wmma_f32_16x16x32_bf16(false, a0, false, b0, (short)0, acc00, false, false);
    acc01 = __builtin_amdgcn_wmma_f32_16x16x32_bf16(false, a0, false, b1, (short)0, acc01, false, false);
    acc10 = __builtin_amdgcn_wmma_f32_16x16x32_bf16(false, a1, false, b0, (short)0, acc10, false, false);
    acc11 = __builtin_amdgcn_wmma_f32_16x16x32_bf16(false, a1, false, b1, (short)0, acc11, false, false);
  }

  const int rowOff = 8 * (lane >> 4);
  auto storeTile = [&](const v8f& acc, int mt, int nt) {
    const int col = n0 + nt * 16 + l15;
    #pragma unroll
    for (int r = 0; r < 8; ++r) {
      const int mrow = m0 + mt * 16 + r + rowOff;
      C[(long long)mrow * ldc + col] = acc[r];
    }
  };
  storeTile(acc00, 0, 0);
  storeTile(acc01, 0, 1);
  storeTile(acc10, 1, 0);
  storeTile(acc11, 1, 1);
}

// ---------------------------------------------------------------------------
// Elementwise / utility kernels
// ---------------------------------------------------------------------------
__global__ void fill_kernel(float* __restrict__ p, float v, long long n) {
  long long i = (long long)blockIdx.x * blockDim.x + threadIdx.x;
  if (i < n) p[i] = v;
}

__global__ void relu_kernel(float* __restrict__ p, long long n) {
  long long i = (long long)blockIdx.x * blockDim.x + threadIdx.x;
  if (i < n) p[i] = fmaxf(p[i], 0.0f);
}

// row-wise softmax (in place, f32) + fused transposed bf16 copy:
//   st[b][f][n] = softmax(X[row])[f]  with b = row/nPerB, n = row%nPerB
// One wave per row; W multiple of 32.
__global__ void softmax_rows_tr(float* __restrict__ X, int M, int W,
                                __bf16* __restrict__ st, int nPerB)
{
  int row = blockIdx.x * (blockDim.x >> 5) + (threadIdx.x >> 5);
  int lane = threadIdx.x & 31;
  if (row >= M) return;
  float* r = X + (long long)row * W;
  int b = row / nPerB, n = row - b * nPerB;
  __bf16* stp = st + (size_t)b * W * nPerB + n;

  float mx = -__builtin_inff();
  for (int f = lane; f < W; f += 32) mx = fmaxf(mx, r[f]);
  #pragma unroll
  for (int o = 16; o > 0; o >>= 1) mx = fmaxf(mx, __shfl_xor(mx, o, 32));
  float s = 0.0f;
  for (int f = lane; f < W; f += 32) { float e = __expf(r[f] - mx); r[f] = e; s += e; }
  s = waveSum(s);
  float inv = 1.0f / s;
  for (int f = lane; f < W; f += 32) {
    float v = r[f] * inv;
    r[f] = v;
    stp[(size_t)f * nPerB] = (__bf16)v;
  }
}

// ---------------------------------------------------------------------------
// TransformerConv edge kernels (one wave per edge; H=128 -> one float4/lane)
// ---------------------------------------------------------------------------
__global__ void edge_score(const float* __restrict__ q, const float* __restrict__ kf,
                           const float* __restrict__ ea, const float* __restrict__ We,
                           const int* __restrict__ src, const int* __restrict__ dst,
                           float* __restrict__ score, float* __restrict__ mmax)
{
  int e = blockIdx.x * (blockDim.x >> 5) + (threadIdx.x >> 5);
  int lane = threadIdx.x & 31;
  if (e >= EE) return;
  int s = src[e], d = dst[e];
  float a0 = ea[e * 5 + 0], a1 = ea[e * 5 + 1], a2 = ea[e * 5 + 2];
  float a3 = ea[e * 5 + 3], a4 = ea[e * 5 + 4];
  float4 w0 = ((const float4*)(We + 0 * HH))[lane];
  float4 w1 = ((const float4*)(We + 1 * HH))[lane];
  float4 w2 = ((const float4*)(We + 2 * HH))[lane];
  float4 w3 = ((const float4*)(We + 3 * HH))[lane];
  float4 w4 = ((const float4*)(We + 4 * HH))[lane];
  float4 q4 = ((const float4*)(q + (long long)d * HH))[lane];
  float4 k4 = ((const float4*)(kf + (long long)s * HH))[lane];
  float ex = a0 * w0.x + a1 * w1.x + a2 * w2.x + a3 * w3.x + a4 * w4.x;
  float ey = a0 * w0.y + a1 * w1.y + a2 * w2.y + a3 * w3.y + a4 * w4.y;
  float ez = a0 * w0.z + a1 * w1.z + a2 * w2.z + a3 * w3.z + a4 * w4.z;
  float ew = a0 * w0.w + a1 * w1.w + a2 * w2.w + a3 * w3.w + a4 * w4.w;
  float acc = q4.x * (k4.x + ex) + q4.y * (k4.y + ey) +
              q4.z * (k4.z + ez) + q4.w * (k4.w + ew);
  acc = waveSum(acc);
  if (lane == 0) {
    float sc = acc * 0.08838834764831845f;   // 1/sqrt(128)
    score[e] = sc;
    atomicMaxFloat(&mmax[d], sc);
  }
}

__global__ void edge_pden(const float* __restrict__ score, const float* __restrict__ mmax,
                          const int* __restrict__ dst,
                          float* __restrict__ pv, float* __restrict__ den)
{
  int e = blockIdx.x * blockDim.x + threadIdx.x;
  if (e >= EE) return;
  int d = dst[e];
  float p = __expf(score[e] - mmax[d]);
  pv[e] = p;
  atomicAdd(&den[d], p);
}

__global__ void edge_agg(const float* __restrict__ v, const float* __restrict__ ea,
                         const float* __restrict__ We,
                         const int* __restrict__ src, const int* __restrict__ dst,
                         const float* __restrict__ pv, const float* __restrict__ den,
                         float* __restrict__ agg)
{
  int e = blockIdx.x * (blockDim.x >> 5) + (threadIdx.x >> 5);
  int lane = threadIdx.x & 31;
  if (e >= EE) return;
  int s = src[e], d = dst[e];
  float alpha = pv[e] / (den[d] + 1e-16f);
  float a0 = ea[e * 5 + 0], a1 = ea[e * 5 + 1], a2 = ea[e * 5 + 2];
  float a3 = ea[e * 5 + 3], a4 = ea[e * 5 + 4];
  float4 w0 = ((const float4*)(We + 0 * HH))[lane];
  float4 w1 = ((const float4*)(We + 1 * HH))[lane];
  float4 w2 = ((const float4*)(We + 2 * HH))[lane];
  float4 w3 = ((const float4*)(We + 3 * HH))[lane];
  float4 w4 = ((const float4*)(We + 4 * HH))[lane];
  float4 v4 = ((const float4*)(v + (long long)s * HH))[lane];
  float* ad = agg + (long long)d * HH + lane * 4;
  float ex = a0 * w0.x + a1 * w1.x + a2 * w2.x + a3 * w3.x + a4 * w4.x;
  float ey = a0 * w0.y + a1 * w1.y + a2 * w2.y + a3 * w3.y + a4 * w4.y;
  float ez = a0 * w0.z + a1 * w1.z + a2 * w2.z + a3 * w3.z + a4 * w4.z;
  float ew = a0 * w0.w + a1 * w1.w + a2 * w2.w + a3 * w3.w + a4 * w4.w;
  atomicAdd(ad + 0, alpha * (v4.x + ex));
  atomicAdd(ad + 1, alpha * (v4.y + ey));
  atomicAdd(ad + 2, alpha * (v4.z + ez));
  atomicAdd(ad + 3, alpha * (v4.w + ew));
}

__global__ void edge_adj(const int* __restrict__ src, const int* __restrict__ dst,
                         float* __restrict__ adj)
{
  int e = blockIdx.x * blockDim.x + threadIdx.x;
  if (e >= EE) return;
  int s = src[e], d = dst[e];
  int b = s >> 9;                 // N = 512
  int ps = s & (NND - 1), pd = d & (NND - 1);
  atomicAdd(&adj[(long long)b * NND * NND + (long long)ps * NND + pd], 1.0f);
}

// ---------------------------------------------------------------------------
// mincut adjacency post-process: zero diagonal, d = sqrt(rowsum)+1e-15,
// then A /= d[i]*d[j]
// ---------------------------------------------------------------------------
__global__ void adj_diag_rowsum(float* __restrict__ adjM, float* __restrict__ dvec,
                                int nb, int kd)
{
  int row = blockIdx.x * (blockDim.x >> 5) + (threadIdx.x >> 5);
  int lane = threadIdx.x & 31;
  if (row >= nb * kd) return;
  int b = row / kd, i = row - b * kd;
  float* r = adjM + (long long)b * kd * kd + (long long)i * kd;
  float s = 0.0f;
  for (int f = lane; f < kd; f += 32) {
    float v = r[f];
    if (f == i) { r[f] = 0.0f; v = 0.0f; }
    s += v;
  }
  s = waveSum(s);
  if (lane == 0) dvec[row] = sqrtf(s) + 1e-15f;
}

__global__ void adj_norm(float* __restrict__ adjM, const float* __restrict__ dvec,
                         int nb, int kd)
{
  long long idx = (long long)blockIdx.x * blockDim.x + threadIdx.x;
  long long tot = (long long)nb * kd * kd;
  if (idx >= tot) return;
  int j = (int)(idx % kd);
  long long t = idx / kd;
  int i = (int)(t % kd);
  int b = (int)(t / kd);
  adjM[idx] = adjM[idx] / (dvec[b * kd + i] * dvec[b * kd + j]);
}

// ---------------------------------------------------------------------------
// head: out[b] = sigmoid( mean_n( x3[b,n,:] @ Wro ) + bro ) ; one wave/graph
// ---------------------------------------------------------------------------
__global__ void head_kernel(const float* __restrict__ x3, const float* __restrict__ Wro,
                            const float* __restrict__ bro, float* __restrict__ out,
                            int nb, int nn, int ff)
{
  int b = blockIdx.x * (blockDim.x >> 5) + (threadIdx.x >> 5);
  int lane = threadIdx.x & 31;
  if (b >= nb) return;
  float acc = 0.0f;
  for (int n = lane; n < nn; n += 32) {
    const float4* r = (const float4*)(x3 + ((long long)b * nn + n) * ff);
    const float4* w = (const float4*)Wro;
    float d = 0.0f;
    #pragma unroll 8
    for (int f = 0; f < ff / 4; ++f) {
      float4 a = r[f], ww = w[f];
      d += a.x * ww.x + a.y * ww.y + a.z * ww.z + a.w * ww.w;
    }
    acc += d;
  }
  acc = waveSum(acc);
  if (lane == 0) {
    float m = acc / (float)nn + bro[0];
    out[b] = 1.0f / (1.0f + __expf(-m));
  }
}

// ---------------------------------------------------------------------------
// kernel_launch
// ---------------------------------------------------------------------------
extern "C" void kernel_launch(void* const* d_in, const int* in_sizes, int n_in,
                              void* d_out, int out_size, void* d_ws, size_t ws_size,
                              hipStream_t stream)
{
  (void)in_sizes; (void)n_in; (void)out_size;
  const float* x     = (const float*)d_in[0];
  const int*   ei    = (const int*)d_in[1];   // [2,E]: src then dst
  const float* ea    = (const float*)d_in[2];
  const float* Wq    = (const float*)d_in[4];  const float* bq    = (const float*)d_in[5];
  const float* Wk    = (const float*)d_in[6];  const float* bk    = (const float*)d_in[7];
  const float* Wv    = (const float*)d_in[8];  const float* bv    = (const float*)d_in[9];
  const float* We    = (const float*)d_in[10];
  const float* Wskip = (const float*)d_in[11]; const float* bskip = (const float*)d_in[12];
  const float* Wm1   = (const float*)d_in[13]; const float* bm1   = (const float*)d_in[14];
  const float* W2r   = (const float*)d_in[15]; const float* b2r   = (const float*)d_in[16];
  const float* W2l   = (const float*)d_in[17];
  const float* Wm2   = (const float*)d_in[18]; const float* bm2   = (const float*)d_in[19];
  const float* W3r   = (const float*)d_in[20]; const float* b3r   = (const float*)d_in[21];
  const float* W3l   = (const float*)d_in[22];
  const float* Wl1   = (const float*)d_in[23]; const float* bl1   = (const float*)d_in[24];
  const float* Wro   = (const float*)d_in[25]; const float* bro   = (const float*)d_in[26];
  float* out = (float*)d_out;
  const int* srcI = ei;
  const int* dstI = ei + EE;

  // ---- workspace carve-out (floats), lifetime-based reuse ----
  float* ws = (float*)d_ws;
  size_t off = 0;
  auto alloc = [&](size_t n) { float* p = ws + off; off += n; return p; };
  float* SCORE = alloc(EE);
  float* PV    = alloc(EE);
  float* MMAX  = alloc(NT);
  float* DEN   = alloc(NT);
  float* QB    = alloc((size_t)NT * HH);   // q ; later s1 (with KB)
  float* KB    = alloc((size_t)NT * HH);   // k ; (s1 tail)
  float* VB    = alloc((size_t)NT * HH);   // v ; later x1 | t1
  float* AGG   = alloc((size_t)NT * HH);   // skip->agg->h ; later s2 | x2 | x3
  float* SA1   = alloc((size_t)BGR * KP1 * NND); // s1^T A ; later u1 | x1n | sa2
  float* ADJ   = alloc((size_t)BGR * NND * NND); // dense A ; later adj1 | d1,d2,adj2,t2,u2,x2n
  float* TRB   = alloc((size_t)BGR * KP1 * NND / 2); // bf16 transposed s1 / s2
  if (off * sizeof(float) > ws_size) return;

  float* S1   = QB;
  float* H_   = AGG;
  float* X1   = VB;
  float* T1   = VB + (size_t)BGR * KP1 * HH;
  float* S2   = AGG;
  float* X2   = AGG + (size_t)BGR * KP1 * KP2;
  float* X3   = X2 + (size_t)BGR * KP2 * HH;
  float* ADJ1 = ADJ;
  float* D1   = ADJ + (size_t)BGR * KP1 * KP1;
  float* D2   = D1 + (size_t)BGR * KP1;
  float* ADJ2 = D2 + (size_t)BGR * KP2;
  float* T2   = ADJ2 + (size_t)BGR * KP2 * KP2;
  float* U2   = T2 + (size_t)BGR * KP2 * HH;
  float* X2N  = U2 + (size_t)BGR * KP2 * HH;
  float* U1   = SA1;
  float* X1N  = SA1 + (size_t)BGR * KP1 * HH;
  float* SA2  = X1N + (size_t)BGR * KP1 * HH;
  __bf16* ST1 = (__bf16*)TRB;              // [B,K1,N] bf16
  __bf16* ST2 = (__bf16*)TRB;              // [B,K2,K1] bf16 (reuse)

  auto gemm = [&](const float* A, const float* Bw, const float* bias, const float* Cadd,
                  float* C, int M, int N, int K,
                  long long sA, long long sB, long long sC,
                  int lda, int ldb, int ldc, int relu, int batch) {
    dim3 g((unsigned)((M + 31) / 32), (unsigned)((N + 127) / 128), (unsigned)batch);
    dim3 blk(128, 1, 1);
    wmma_gemm_f32a_kernel<<<g, blk, 0, stream>>>(A, Bw, bias, Cadd, C, M, N, K,
                                                 sA, sB, sC, lda, ldb, ldc, relu);
  };
  auto gemmT = [&](const __bf16* A, const float* Bw, float* C, int M, int N, int K,
                   long long sA, long long sB, long long sC,
                   int lda, int ldb, int ldc, int batch) {
    dim3 g((unsigned)((M + 31) / 32), (unsigned)((N + 127) / 128), (unsigned)batch);
    dim3 blk(128, 1, 1);
    wmma_gemm_bf16a_kernel<<<g, blk, 0, stream>>>(A, Bw, C, M, N, K,
                                                  sA, sB, sC, lda, ldb, ldc);
  };

  // ---- init buffers ----
  fill_kernel<<<(NT + 255) / 256, 256, 0, stream>>>(MMAX, -__builtin_inff(), NT);
  fill_kernel<<<(NT + 255) / 256, 256, 0, stream>>>(DEN, 0.0f, NT);
  {
    long long n = (long long)BGR * NND * NND;
    fill_kernel<<<(unsigned)((n + 255) / 256), 256, 0, stream>>>(ADJ, 0.0f, n);
  }

  // ---- TransformerConv ----
  gemm(x, Wq,    bq,    nullptr, QB,  NT, HH, FIN, 0, 0, 0, FIN, HH, HH, 0, 1);
  gemm(x, Wk,    bk,    nullptr, KB,  NT, HH, FIN, 0, 0, 0, FIN, HH, HH, 0, 1);
  gemm(x, Wv,    bv,    nullptr, VB,  NT, HH, FIN, 0, 0, 0, FIN, HH, HH, 0, 1);
  gemm(x, Wskip, bskip, nullptr, AGG, NT, HH, FIN, 0, 0, 0, FIN, HH, HH, 0, 1);

  edge_score<<<(EE + 7) / 8, 256, 0, stream>>>(QB, KB, ea, We, srcI, dstI, SCORE, MMAX);
  edge_pden<<<(EE + 255) / 256, 256, 0, stream>>>(SCORE, MMAX, dstI, PV, DEN);
  edge_agg<<<(EE + 7) / 8, 256, 0, stream>>>(VB, ea, We, srcI, dstI, PV, DEN, AGG);
  edge_adj<<<(EE + 255) / 256, 256, 0, stream>>>(srcI, dstI, ADJ);
  relu_kernel<<<(unsigned)(((long long)NT * HH + 255) / 256), 256, 0, stream>>>(AGG, (long long)NT * HH);

  // ---- pool 1 ----
  gemm(H_, Wm1, bm1, nullptr, S1, NT, KP1, HH, 0, 0, 0, HH, KP1, KP1, 0, 1);
  softmax_rows_tr<<<(NT + 7) / 8, 256, 0, stream>>>(S1, NT, KP1, ST1, NND);
  // x1 = s1^T @ h
  gemmT(ST1, H_, X1, KP1, HH, NND,
        (long long)KP1 * NND, (long long)NND * HH, (long long)KP1 * HH,
        NND, HH, HH, BGR);
  // sa1 = s1^T @ adj
  gemmT(ST1, ADJ, SA1, KP1, NND, NND,
        (long long)KP1 * NND, (long long)NND * NND, (long long)KP1 * NND,
        NND, NND, NND, BGR);
  // adj1 = sa1 @ s1  (overwrites ADJ region)
  gemm(SA1, S1, nullptr, nullptr, ADJ1, KP1, KP1, NND,
       (long long)KP1 * NND, (long long)NND * KP1, (long long)KP1 * KP1,
       NND, KP1, KP1, 0, BGR);
  adj_diag_rowsum<<<(BGR * KP1 + 7) / 8, 256, 0, stream>>>(ADJ1, D1, BGR, KP1);
  {
    long long n = (long long)BGR * KP1 * KP1;
    adj_norm<<<(unsigned)((n + 255) / 256), 256, 0, stream>>>(ADJ1, D1, BGR, KP1);
  }

  // ---- DenseGraphConv 2 + relu ----
  gemm(ADJ1, X1, nullptr, nullptr, T1, KP1, HH, KP1,
       (long long)KP1 * KP1, (long long)KP1 * HH, (long long)KP1 * HH,
       KP1, HH, HH, 0, BGR);
  gemm(X1, W2l, nullptr, nullptr, U1, BGR * KP1, HH, HH, 0, 0, 0, HH, HH, HH, 0, 1);
  gemm(T1, W2r, b2r, U1, X1N, BGR * KP1, HH, HH, 0, 0, 0, HH, HH, HH, 1, 1);

  // ---- pool 2 ----
  gemm(X1N, Wm2, bm2, nullptr, S2, BGR * KP1, KP2, HH, 0, 0, 0, HH, KP2, KP2, 0, 1);
  softmax_rows_tr<<<(BGR * KP1 + 7) / 8, 256, 0, stream>>>(S2, BGR * KP1, KP2, ST2, KP1);
  gemmT(ST2, X1N, X2, KP2, HH, KP1,
        (long long)KP2 * KP1, (long long)KP1 * HH, (long long)KP2 * HH,
        KP1, HH, HH, BGR);
  gemmT(ST2, ADJ1, SA2, KP2, KP1, KP1,
        (long long)KP2 * KP1, (long long)KP1 * KP1, (long long)KP2 * KP1,
        KP1, KP1, KP1, BGR);
  gemm(SA2, S2, nullptr, nullptr, ADJ2, KP2, KP2, KP1,
       (long long)KP2 * KP1, (long long)KP1 * KP2, (long long)KP2 * KP2,
       KP1, KP2, KP2, 0, BGR);
  adj_diag_rowsum<<<(BGR * KP2 + 7) / 8, 256, 0, stream>>>(ADJ2, D2, BGR, KP2);
  {
    long long n = (long long)BGR * KP2 * KP2;
    adj_norm<<<(unsigned)((n + 255) / 256), 256, 0, stream>>>(ADJ2, D2, BGR, KP2);
  }

  // ---- DenseGraphConv 3 (no relu) + head MLP ----
  gemm(ADJ2, X2, nullptr, nullptr, T2, KP2, HH, KP2,
       (long long)KP2 * KP2, (long long)KP2 * HH, (long long)KP2 * HH,
       KP2, HH, HH, 0, BGR);
  gemm(X2, W3l, nullptr, nullptr, U2, BGR * KP2, HH, HH, 0, 0, 0, HH, HH, HH, 0, 1);
  gemm(T2, W3r, b3r, U2, X2N, BGR * KP2, HH, HH, 0, 0, 0, HH, HH, HH, 0, 1);
  gemm(X2N, Wl1, bl1, nullptr, X3, BGR * KP2, HH, HH, 0, 0, 0, HH, HH, HH, 1, 1);

  head_kernel<<<(BGR + 7) / 8, 256, 0, stream>>>(X3, Wro, bro, out, BGR, KP2, HH);
}
